// MultiHeadAttention_38620345925806
// MI455X (gfx1250) — compile-verified, matrix operand order checked by reading, not run on hardware
//
#include <hip/hip_runtime.h>
#include <hip/hip_bf16.h>
#include <stdint.h>

// Problem constants (match reference)
#define BB 2
#define LL 2048
#define DD 1024
#define HH 16
#define DKK 64

typedef __bf16 bf16;
typedef bf16  v16bf __attribute__((ext_vector_type(16)));
typedef bf16  v8bf  __attribute__((ext_vector_type(8)));
typedef float v8f   __attribute__((ext_vector_type(8)));

__device__ __forceinline__ v16bf cat8(v8bf lo, v8bf hi) {
  return __builtin_shufflevector(lo, hi, 0,1,2,3,4,5,6,7,8,9,10,11,12,13,14,15);
}

// DPP16 ROW_XMASK lane swap within rows of 16 lanes (matches the C/D-matrix
// half-wave groups). VALU-only: no LDS traffic, unlike ds_bpermute shuffles.
template <int XMASK>
__device__ __forceinline__ float dpp_xor(float v) {
  int s = __builtin_amdgcn_update_dpp(0, __builtin_bit_cast(int, v),
                                      0x160 | XMASK, 0xf, 0xf, true);
  return __builtin_bit_cast(float, s);
}
__device__ __forceinline__ float row16_max(float v) {
  v = fmaxf(v, dpp_xor<1>(v));
  v = fmaxf(v, dpp_xor<2>(v));
  v = fmaxf(v, dpp_xor<4>(v));
  v = fmaxf(v, dpp_xor<8>(v));
  return v;
}
__device__ __forceinline__ float row16_sum(float v) {
  v += dpp_xor<1>(v);
  v += dpp_xor<2>(v);
  v += dpp_xor<4>(v);
  v += dpp_xor<8>(v);
  return v;
}

// Load one 16x32 bf16 WMMA fragment (A or B side; both use the same per-lane
// pattern: row = lane&15, K = kbase + 8*(lane>>4) + {0..7, 16..23}).
// strideE = LDS row stride in elements (multiple of 8 -> ds_load_b128 aligned).
__device__ __forceinline__ v16bf load_frag(const bf16* lds, int strideE, int row, int kbase) {
  const int lane = threadIdx.x & 31;
  const bf16* p = lds + (row + (lane & 15)) * strideE + kbase + ((lane >> 4) << 3);
  v8bf lo = *(const v8bf*)(p);
  v8bf hi = *(const v8bf*)(p + 16);
  return cat8(lo, hi);
}

// ---------------------------------------------------------------------------
// GEMM: Y = A[M,K] @ W[N,K]^T + bias[N], then Y *= outScale.
//   MODE 0: A=f32, out bf16 [B,H,L,DK]   (q/k projection, head-split)
//   MODE 1: A=f32, out bf16 [B,H,DK,L]   (v projection, pre-transposed)
//   MODE 2: A=bf16, out f32 [M,N]        (final fc)
// M=4096, N=K=1024. Block tile 128x128, K-step 32, 8 waves (16-row strip each,
// 8 accumulators wide). Double-buffered LDS: one barrier per K-step, global
// loads for step k+1 overlap WMMAs for step k.
// ---------------------------------------------------------------------------
template <int MODE>
__global__ __launch_bounds__(256) void mha_gemm_wmma(
    const void* __restrict__ Ain, const float* __restrict__ W,
    const float* __restrict__ bias, void* __restrict__ Out, float outScale) {
  constexpr int N = DD, K = DD;
  constexpr int BM = 128, BN = 128, BK = 32;
  constexpr int AS = 40, BS = 40;  // LDS row strides (elements) -> 80B, 16B-aligned
  __shared__ bf16 sA[2][BM * AS];
  __shared__ bf16 sB[2][BN * BS];

  const int tid  = threadIdx.x;
  const int wave = tid >> 5;
  const int lane = tid & 31;
  const int m0 = blockIdx.x * BM;
  const int n0 = blockIdx.y * BN;

  v8f acc[8] = {};

  // ---- staging registers for the in-flight global tile ----
  float4 rB[4];            // W tile: 128 rows x 8 chunks of 4 f32 -> 4/thread
  float4 rAf[4];           // A tile (f32 modes): 128 x 8 chunks of 4 f32
  v8bf   rAh[2];           // A tile (bf16 mode): 128 x 4 chunks of 8 bf16

  auto load_tile = [&](int k0) {
    if constexpr (MODE == 2) {
      const bf16* A = (const bf16*)Ain;
#pragma unroll
      for (int p = 0; p < 2; ++p) {
        int idx = p * 256 + tid;
        int r = idx >> 2, c = (idx & 3) << 3;
        rAh[p] = *(const v8bf*)(A + (size_t)(m0 + r) * K + k0 + c);
      }
    } else {
      const float* A = (const float*)Ain;
#pragma unroll
      for (int p = 0; p < 4; ++p) {
        int idx = p * 256 + tid;
        int r = idx >> 3, c = (idx & 7) << 2;
        rAf[p] = *(const float4*)(A + (size_t)(m0 + r) * K + k0 + c);
      }
    }
#pragma unroll
    for (int p = 0; p < 4; ++p) {
      int idx = p * 256 + tid;
      int r = idx >> 3, c = (idx & 7) << 2;
      rB[p] = *(const float4*)(W + (size_t)(n0 + r) * K + k0 + c);
    }
  };

  auto store_tile = [&](int buf) {
    if constexpr (MODE == 2) {
#pragma unroll
      for (int p = 0; p < 2; ++p) {
        int idx = p * 256 + tid;
        int r = idx >> 2, c = (idx & 3) << 3;
        *(v8bf*)(sA[buf] + r * AS + c) = rAh[p];
      }
    } else {
#pragma unroll
      for (int p = 0; p < 4; ++p) {
        int idx = p * 256 + tid;
        int r = idx >> 3, c = (idx & 7) << 2;
        bf16* d = sA[buf] + r * AS + c;
        d[0] = (bf16)rAf[p].x; d[1] = (bf16)rAf[p].y;
        d[2] = (bf16)rAf[p].z; d[3] = (bf16)rAf[p].w;
      }
    }
#pragma unroll
    for (int p = 0; p < 4; ++p) {
      int idx = p * 256 + tid;
      int r = idx >> 3, c = (idx & 7) << 2;
      bf16* d = sB[buf] + r * BS + c;
      d[0] = (bf16)rB[p].x; d[1] = (bf16)rB[p].y;
      d[2] = (bf16)rB[p].z; d[3] = (bf16)rB[p].w;
    }
  };

  load_tile(0);
  int buf = 0;
  for (int k0 = 0; k0 < K; k0 += BK) {
    store_tile(buf);
    __syncthreads();
    if (k0 + BK < K) load_tile(k0 + BK);
    if (k0 + 2 * BK < K) {  // prefetch two tiles ahead into L2/WGP$
      if constexpr (MODE != 2)
        __builtin_prefetch((const float*)Ain + (size_t)(m0 + (tid >> 1)) * K + k0 + 2 * BK, 0, 3);
      __builtin_prefetch(W + (size_t)(n0 + (tid >> 1)) * K + k0 + 2 * BK, 0, 3);
    }
    v16bf a = load_frag(sA[buf], AS, wave * 16, 0);
#pragma unroll
    for (int nt = 0; nt < 8; ++nt) {
      v16bf b = load_frag(sB[buf], BS, nt * 16, 0);
      acc[nt] = __builtin_amdgcn_wmma_f32_16x16x32_bf16(
          false, a, false, b, (short)0, acc[nt], false, false);
    }
    buf ^= 1;
  }

  // ---- epilogue: bias + scale + scatter store per output mode ----
#pragma unroll
  for (int nt = 0; nt < 8; ++nt) {
    const int col = n0 + nt * 16 + (lane & 15);
    const float bv = bias[col];
#pragma unroll
    for (int r = 0; r < 8; ++r) {
      const int m = m0 + wave * 16 + ((lane >> 4) << 3) + r;
      const float v = (acc[nt][r] + bv) * outScale;
      if constexpr (MODE == 2) {
        ((float*)Out)[(size_t)m * N + col] = v;
      } else {
        const int b = m >> 11, l = m & (LL - 1);
        const int h = col >> 6, dk = col & (DKK - 1);
        const size_t off = (MODE == 0)
            ? ((size_t)((b * HH + h) * LL + l) * DKK + dk)    // [B,H,L,DK]
            : ((size_t)((b * HH + h) * DKK + dk) * LL + l);   // [B,H,DK,L]
        ((bf16*)Out)[off] = (bf16)v;
      }
    }
  }
}

// ---------------------------------------------------------------------------
// Flash attention (causal). grid = (L/128, B*H), block = 256 (8 waves x 16 q-rows).
// Q: bf16 [B,H,L,DK] pre-scaled by 1/sqrt(DK); K: bf16 [B,H,L,DK];
// Vt: bf16 [B,H,DK,L]; out: bf16 [B,L,D] (heads merged).
// Double-buffered K/V tiles: one barrier per key-tile.
// ---------------------------------------------------------------------------
__global__ __launch_bounds__(256) void mha_attn_wmma(
    const bf16* __restrict__ Qp, const bf16* __restrict__ Kp,
    const bf16* __restrict__ Vt, bf16* __restrict__ Op) {
  constexpr int KT = 64;   // keys per tile
  constexpr int QS = 128;  // queries per block
  constexpr int LS = 72;   // LDS row stride (elements) -> 144B, 16B-aligned
  __shared__ bf16 sK[2][KT * LS];   // [key][dk]
  __shared__ bf16 sV[2][DKK * LS];  // [dk][key]  (V pre-transposed in global)
  __shared__ bf16 sP[QS * LS];      // [qrow][key], wave-private 16-row strips

  const int tid  = threadIdx.x;
  const int wave = tid >> 5;
  const int lane = tid & 31;
  const int qt = blockIdx.x;
  const int bh = blockIdx.y;
  const int q0 = qt * QS;

  const bf16* Qb = Qp + (size_t)bh * LL * DKK;
  const bf16* Kb = Kp + (size_t)bh * LL * DKK;
  const bf16* Vb = Vt + (size_t)bh * DKK * LL;

  // Load this wave's Q A-fragments (16 rows x 64 features) once from global.
  v16bf qf[2];
  {
    const int qrow = q0 + wave * 16 + (lane & 15);
    const int kh = (lane >> 4) << 3;
    const bf16* p = Qb + (size_t)qrow * DKK;
#pragma unroll
    for (int c = 0; c < 2; ++c) {
      v8bf lo = *(const v8bf*)(p + c * 32 + kh);
      v8bf hi = *(const v8bf*)(p + c * 32 + kh + 16);
      qf[c] = cat8(lo, hi);
    }
  }

  // Tile staging registers (2 x uint4 each for K and V).
  uint4 rk[2], rv[2];
  auto load_kv = [&](int kt) {
    const uint4* gk = (const uint4*)(Kb + (size_t)kt * KT * DKK);  // contiguous tile
#pragma unroll
    for (int p = 0; p < 2; ++p) {
      int idx = p * 256 + tid;          // 512 x uint4 per tile
      int r = idx >> 3, c = idx & 7;    // 8 x uint4 per 64-elem row
      rk[p] = gk[idx];
      rv[p] = *(const uint4*)(Vb + (size_t)r * LL + kt * KT + (c << 3));
    }
  };
  auto store_kv = [&](int buf) {
#pragma unroll
    for (int p = 0; p < 2; ++p) {
      int idx = p * 256 + tid;
      int r = idx >> 3, c = idx & 7;
      *(uint4*)(sK[buf] + r * LS + (c << 3)) = rk[p];
      *(uint4*)(sV[buf] + r * LS + (c << 3)) = rv[p];
    }
  };

  float mrow[8], lrow[8];
  v8f o[4] = {};
#pragma unroll
  for (int r = 0; r < 8; ++r) { mrow[r] = -1e30f; lrow[r] = 0.f; }

  const int nkt = (q0 + QS) / KT;  // causal: only tiles at/below the diagonal
  load_kv(0);
  int buf = 0;
  for (int kt = 0; kt < nkt; ++kt) {
    store_kv(buf);
    __syncthreads();
    if (kt + 1 < nkt) load_kv(kt + 1);

    // ---- S = q @ K^T  (16 x 64 per wave; scale pre-folded into q) ----
    v8f s[4] = {};
#pragma unroll
    for (int nt = 0; nt < 4; ++nt) {
#pragma unroll
      for (int c = 0; c < 2; ++c) {
        v16bf b = load_frag(sK[buf], LS, nt * 16, c * 32);
        s[nt] = __builtin_amdgcn_wmma_f32_16x16x32_bf16(
            false, qf[c], false, b, (short)0, s[nt], false, false);
      }
    }

    // ---- causal mask (skipped for wave-uniformly unmasked tiles) ----
    const int qbase = q0 + wave * 16 + ((lane >> 4) << 3);
    const bool needMask = (kt * KT + KT - 1) > (q0 + wave * 16);  // wave-uniform
    if (needMask) {
      const int kcol = kt * KT + (lane & 15);
#pragma unroll
      for (int r = 0; r < 8; ++r)
#pragma unroll
        for (int nt = 0; nt < 4; ++nt)
          s[nt][r] = (kcol + nt * 16 <= qbase + r) ? s[nt][r] : -1e30f;
    }

    // ---- online softmax (DPP16 butterflies across the 16-lane row groups) ----
#pragma unroll
    for (int r = 0; r < 8; ++r) {
      float mx = fmaxf(fmaxf(s[0][r], s[1][r]), fmaxf(s[2][r], s[3][r]));
      mx = row16_max(mx);
      const float newm = fmaxf(mrow[r], mx);
      const float corr = __expf(mrow[r] - newm);
      mrow[r] = newm;
      float rsum = 0.f;
#pragma unroll
      for (int nt = 0; nt < 4; ++nt) {
        const float p = __expf(s[nt][r] - newm);
        s[nt][r] = p;
        rsum += p;
      }
      rsum = row16_sum(rsum);
      lrow[r] = lrow[r] * corr + rsum;
#pragma unroll
      for (int nt = 0; nt < 4; ++nt) o[nt][r] *= corr;
    }

    // ---- stage P (wave-private LDS strip; wave-internal DS ordering, no barrier) ----
    {
      bf16* pw = sP + wave * 16 * LS;
#pragma unroll
      for (int nt = 0; nt < 4; ++nt)
#pragma unroll
        for (int r = 0; r < 8; ++r)
          pw[(((lane >> 4) << 3) + r) * LS + nt * 16 + (lane & 15)] = (bf16)s[nt][r];
    }

    // ---- O += P @ V ----
#pragma unroll
    for (int c = 0; c < 2; ++c) {
      v16bf a = load_frag(sP, LS, wave * 16, c * 32);
#pragma unroll
      for (int nt = 0; nt < 4; ++nt) {
        v16bf b = load_frag(sV[buf], LS, nt * 16, c * 32);
        o[nt] = __builtin_amdgcn_wmma_f32_16x16x32_bf16(
            false, a, false, b, (short)0, o[nt], false, false);
      }
    }
    buf ^= 1;
  }

  // ---- normalize + store merged-head output bf16 [B, L, D] ----
  const int b = bh >> 4, h = bh & (HH - 1);
#pragma unroll
  for (int r = 0; r < 8; ++r) {
    const float inv = 1.0f / lrow[r];
    const int q = q0 + wave * 16 + ((lane >> 4) << 3) + r;
    bf16* dst = Op + ((size_t)(b * LL + q)) * DD + h * DKK + (lane & 15);
#pragma unroll
    for (int nt = 0; nt < 4; ++nt) dst[nt * 16] = (bf16)(o[nt][r] * inv);
  }
}

// ---------------------------------------------------------------------------
extern "C" void kernel_launch(void* const* d_in, const int* in_sizes, int n_in,
                              void* d_out, int out_size, void* d_ws, size_t ws_size,
                              hipStream_t stream) {
  (void)in_sizes; (void)n_in; (void)out_size; (void)ws_size;
  const float* Q    = (const float*)d_in[0];
  const float* K    = (const float*)d_in[1];
  const float* V    = (const float*)d_in[2];
  // d_in[3] = causal mask: applied analytically, not read.
  const float* Wq_w = (const float*)d_in[4];
  const float* Wq_b = (const float*)d_in[5];
  const float* Wk_w = (const float*)d_in[6];
  const float* Wk_b = (const float*)d_in[7];
  const float* Wv_w = (const float*)d_in[8];
  const float* Wv_b = (const float*)d_in[9];
  const float* fc_w = (const float*)d_in[10];
  const float* fc_b = (const float*)d_in[11];

  const size_t elems = (size_t)BB * HH * LL * DKK;  // 4M bf16 each
  bf16* qws = (bf16*)d_ws;
  bf16* kws = qws + elems;
  bf16* vws = kws + elems;  // stored [B,H,DK,L]
  bf16* aws = vws + elems;  // attn output, [B,L,D]

  const dim3 gG(BB * LL / 128, DD / 128);  // 32 x 8
  const float qscale = 0.125f;             // 1/sqrt(DK), folded into q projection
  mha_gemm_wmma<0><<<gG, 256, 0, stream>>>(Q, Wq_w, Wq_b, qws, qscale);
  mha_gemm_wmma<0><<<gG, 256, 0, stream>>>(K, Wk_w, Wk_b, kws, 1.0f);
  mha_gemm_wmma<1><<<gG, 256, 0, stream>>>(V, Wv_w, Wv_b, vws, 1.0f);

  mha_attn_wmma<<<dim3(LL / 128, BB * HH), 256, 0, stream>>>(qws, kws, vws, aws);

  mha_gemm_wmma<2><<<gG, 256, 0, stream>>>(aws, fc_w, fc_b, d_out, 1.0f);
}
